// LUA2Net_84112639525484
// MI455X (gfx1250) — matrix operand order repaired and checked by reading, not compiled, not run on hardware
//
#include <hip/hip_runtime.h>
#include <math.h>

typedef __attribute__((ext_vector_type(16))) _Float16 v16h;
typedef __attribute__((ext_vector_type(8)))  _Float16 v8h;
typedef __attribute__((ext_vector_type(8)))  float    v8f;

#define EPSN 1e-5f

// ================= pack activations: f32 NCHW (concat up to 3, opt relu)
//                   -> f16 [b][p][Kpad] zero-padded, LDS transpose =================
// grid: (HW/64, Kpad/32, B), block 256
__global__ void pack_act_kernel(const float* __restrict__ i0, int C0,
                                const float* __restrict__ i1, int C1,
                                const float* __restrict__ i2, int C2,
                                _Float16* __restrict__ out,
                                int K, int Kpad, int HW, int relu)
{
    __shared__ _Float16 tile[32][66];
    const int b = blockIdx.z;
    const int kBase = blockIdx.y * 32;
    const int pBase = blockIdx.x * 64;
    const int t = threadIdx.x;

    const float* b0 = i0 + (size_t)b * C0 * HW;
    const float* b1 = i1 ? i1 + (size_t)b * C1 * HW : (const float*)0;
    const float* b2 = i2 ? i2 + (size_t)b * C2 * HW : (const float*)0;

    const int p  = pBase + (t & 63);
    const int kq = (t >> 6) * 8;          // 0,8,16,24
#pragma unroll
    for (int r = 0; r < 8; ++r) {
        int kLocal = kq + r;
        int kk = kBase + kLocal;
        float v = 0.f;
        if (kk < C0)                 v = b0[(size_t)kk * HW + p];
        else if (kk - C0 < C1)       v = b1[(size_t)(kk - C0) * HW + p];
        else if (kk - C0 - C1 < C2)  v = b2[(size_t)(kk - C0 - C1) * HW + p];
        if (relu) v = fmaxf(v, 0.f);
        tile[kLocal][t & 63] = (_Float16)v;
    }
    __syncthreads();
    const int kL = t & 31;
    const int pL = t >> 5;                // 0..7
    _Float16* ob = out + ((size_t)b * HW + pBase) * Kpad + kBase + kL;
#pragma unroll
    for (int r = 0; r < 8; ++r) {
        int p2 = pL * 8 + r;
        ob[(size_t)p2 * Kpad] = tile[kL][p2];
    }
}

// weights (Cout x K f32) -> f16 [Mpad][Kpad] zero padded
__global__ void pack_w_kernel(const float* __restrict__ w, _Float16* __restrict__ out,
                              int Cout, int K, int Kpad, int total)
{
    int i = blockIdx.x * blockDim.x + threadIdx.x;
    if (i >= total) return;
    int m = i / Kpad, k = i % Kpad;
    float v = (m < Cout && k < K) ? w[(size_t)m * K + k] : 0.f;
    out[i] = (_Float16)v;
}

// ================= WMMA GEMM on packed operands =================
// out[b,co,p] = bias[co] + sum_k Wpk[co,k] * Bpk[b,p,k]
// One wave: 16(M) x 32(N = two 16-tiles sharing A). HW % 256 == 0 assumed.
__global__ void gemm_packed_kernel(const _Float16* __restrict__ Apk,
                                   const _Float16* __restrict__ Bpk,
                                   const float* __restrict__ bias,
                                   const float* __restrict__ resid,
                                   float* __restrict__ out,
                                   int Cout, int HW, int Kpad, int relu_out)
{
    const int b    = blockIdx.z;
    const int lane = threadIdx.x & 31;
    const int wv   = threadIdx.x >> 5;
    const int nBase = (blockIdx.x * (blockDim.x >> 5) + wv) * 32;
    const int mBase = blockIdx.y * 16;
    const int ln = lane & 15;
    const int hi = lane >> 4;
    const int p0 = nBase + ln;
    const int p1 = p0 + 16;

    const _Float16* wp = Apk + (size_t)(mBase + ln) * Kpad;        // A row (Mpad padded)
    const _Float16* bp0 = Bpk + ((size_t)b * HW + p0) * Kpad + 16 * hi;
    const _Float16* bp1 = Bpk + ((size_t)b * HW + p1) * Kpad + 16 * hi;

    v8f acc0 = {};
    v8f acc1 = {};
    for (int k0 = 0; k0 < Kpad; k0 += 32) {
        // A fragment: 16 contiguous K halves per lane (K = k0 + 16*hi + 0..15)
        v16h Af = *(const v16h*)(wp + k0 + 16 * hi);
        // B fragments: 16 contiguous K halves per lane (K = k0 + 16*hi + 0..15)
        v16h Bf0 = *(const v16h*)(bp0 + k0);
        v16h Bf1 = *(const v16h*)(bp1 + k0);
        acc0 = __builtin_amdgcn_wmma_f32_16x16x32_f16(false, Af, false, Bf0,
                                                      (short)0, acc0, false, false);
        acc1 = __builtin_amdgcn_wmma_f32_16x16x32_f16(false, Af, false, Bf1,
                                                      (short)0, acc1, false, false);
    }

    float* ob = out + (size_t)b * Cout * HW;
    const float* rb = resid ? resid + (size_t)b * Cout * HW : (const float*)0;
#pragma unroll
    for (int r = 0; r < 8; ++r) {
        int co = mBase + r + 8 * hi;
        if (co < Cout) {
            float v0 = acc0[r] + bias[co];
            float v1 = acc1[r] + bias[co];
            if (rb) {
                v0 += rb[(size_t)co * HW + p0];
                v1 += rb[(size_t)co * HW + p1];
            }
            if (relu_out) { v0 = fmaxf(v0, 0.f); v1 = fmaxf(v1, 0.f); }
            ob[(size_t)co * HW + p0] = v0;
            ob[(size_t)co * HW + p1] = v1;
        }
    }
}

// ================= direct 3x3 conv (3->24), pad 1 =================
__global__ void conv3x3_kernel(const float* x, const float* w, const float* bias,
                               float* y, int Cin, int Cout, int H, int W, int total)
{
    int i = blockIdx.x * blockDim.x + threadIdx.x;
    if (i >= total) return;
    int chw = Cout * H * W;
    int b   = i / chw;
    int rem = i % chw;
    int co  = rem / (H * W);
    int pp  = rem % (H * W);
    int yy  = pp / W, xx = pp % W;
    const float* xb = x + (size_t)b * Cin * H * W;
    float a = bias[co];
    for (int ci = 0; ci < Cin; ++ci) {
        const float* xc = xb + (size_t)ci * H * W;
        const float* wc = w + ((size_t)co * Cin + ci) * 9;
#pragma unroll
        for (int dy = -1; dy <= 1; ++dy) {
            int ys = yy + dy;
            if (ys < 0 || ys >= H) continue;
#pragma unroll
            for (int dx = -1; dx <= 1; ++dx) {
                int xs = xx + dx;
                if (xs < 0 || xs >= W) continue;
                a += xc[(size_t)ys * W + xs] * wc[(dy + 1) * 3 + (dx + 1)];
            }
        }
    }
    y[i] = a;
}

// ================= axial depthwise: y = relu(hconv + vconv + x) =================
__global__ void axial_dw_kernel(const float* x, const float* hw_, const float* hb,
                                const float* vw_, const float* vb, float* y,
                                int C, int H, int W, int k, int total)
{
    int i = blockIdx.x * blockDim.x + threadIdx.x;
    if (i >= total) return;
    int bc  = i / (H * W);
    int c   = bc % C;
    int rem = i % (H * W);
    int yy  = rem / W, xx = rem % W;
    const float* p = x + (size_t)bc * H * W;
    int r = k / 2;
    float h = hb[c];
    for (int t = 0; t < k; ++t) {
        int xs = xx - r + t;
        if (xs >= 0 && xs < W) h += p[(size_t)yy * W + xs] * hw_[c * k + t];
    }
    float v = vb[c];
    for (int t = 0; t < k; ++t) {
        int ys = yy - r + t;
        if (ys >= 0 && ys < H) v += p[(size_t)ys * W + xx] * vw_[c * k + t];
    }
    y[i] = fmaxf(h + v + p[rem], 0.f);
}

// ================= per-(b,c) sum / sumsq reduction =================
__global__ void reduce_stats_kernel(const float* x, float* S1, float* S2, int HW)
{
    const float* p = x + (size_t)blockIdx.x * HW;
    float s1 = 0.f, s2 = 0.f;
    for (int i = threadIdx.x; i < HW; i += blockDim.x) {
        float v = p[i]; s1 += v; s2 += v * v;
    }
    __shared__ float a1[256], a2[256];
    a1[threadIdx.x] = s1; a2[threadIdx.x] = s2;
    __syncthreads();
    for (int s = 128; s > 0; s >>= 1) {
        if (threadIdx.x < s) { a1[threadIdx.x] += a1[threadIdx.x + s]; a2[threadIdx.x] += a2[threadIdx.x + s]; }
        __syncthreads();
    }
    if (threadIdx.x == 0) { S1[blockIdx.x] = a1[0]; S2[blockIdx.x] = a2[0]; }
}

__global__ void batch_stats_kernel(const float* S1, const float* S2,
                                   float* mB, float* vB, int Bn, int C, int HW)
{
    int c = threadIdx.x;
    if (c >= C) return;
    float s1 = 0.f, s2 = 0.f;
    for (int b = 0; b < Bn; ++b) { s1 += S1[b * C + c]; s2 += S2[b * C + c]; }
    float inv = 1.f / ((float)Bn * (float)HW);
    float m = s1 * inv;
    mB[c] = m;
    vB[c] = s2 * inv - m * m;
}

// hnorm: 0.7*instnorm + 0.3*batchnorm(affine)
__global__ void hnorm_apply_kernel(const float* x, float* y,
                                   const float* S1, const float* S2,
                                   const float* mB, const float* vB,
                                   const float* gw, const float* gb,
                                   int C, int HW, int total)
{
    int i = blockIdx.x * blockDim.x + threadIdx.x;
    if (i >= total) return;
    int bc = i / HW;
    int c  = bc % C;
    float invHW = 1.f / (float)HW;
    float m = S1[bc] * invHW;
    float v = S2[bc] * invHW - m * m;
    float xv = x[i];
    float inorm = (xv - m) * rsqrtf(v + EPSN);
    float bn = (xv - mB[c]) * rsqrtf(vB[c] + EPSN) * gw[c] + gb[c];
    y[i] = 0.7f * inorm + 0.3f * bn;
}

__global__ void bn_apply_kernel(const float* x, float* y,
                                const float* mB, const float* vB,
                                const float* gw, const float* gb,
                                int C, int HW, int total, int relu)
{
    int i = blockIdx.x * blockDim.x + threadIdx.x;
    if (i >= total) return;
    int c = (i / HW) % C;
    float v = (x[i] - mB[c]) * rsqrtf(vB[c] + EPSN) * gw[c] + gb[c];
    if (relu) v = fmaxf(v, 0.f);
    y[i] = v;
}

// ================= squeeze-excite =================
__global__ void ca_compute_kernel(const float* S1, const float* W1, const float* W2,
                                  float* s, int C, int Ch, int HW)
{
    int b = blockIdx.x;
    __shared__ float wv[128];
    __shared__ float hid[32];
    int t = threadIdx.x;
    if (t < C) wv[t] = S1[b * C + t] / (float)HW;
    __syncthreads();
    if (t < Ch) {
        float a = 0.f;
        for (int c = 0; c < C; ++c) a += wv[c] * W1[t * C + c];
        hid[t] = fmaxf(a, 0.f);
    }
    __syncthreads();
    if (t < C) {
        float a = 0.f;
        for (int h = 0; h < Ch; ++h) a += hid[h] * W2[t * Ch + h];
        s[b * C + t] = 1.f / (1.f + expf(-a));
    }
}

__global__ void ca_scale_kernel(const float* x, float* y, const float* s, int HW, int total)
{
    int i = blockIdx.x * blockDim.x + threadIdx.x;
    if (i >= total) return;
    y[i] = x[i] * s[i / HW];
}

// ================= maxpool 2x2 =================
__global__ void maxpool2_kernel(const float* x, float* y, int H, int W, int total)
{
    int i = blockIdx.x * blockDim.x + threadIdx.x;
    if (i >= total) return;
    int H2 = H >> 1, W2 = W >> 1;
    int bc  = i / (H2 * W2);
    int rem = i % (H2 * W2);
    int oy = rem / W2, ox = rem % W2;
    const float* p = x + ((size_t)bc * H + 2 * oy) * W + 2 * ox;
    float a = fmaxf(p[0], p[1]);
    float b = fmaxf(p[W], p[W + 1]);
    y[i] = fmaxf(a, b);
}

// ================= bilinear 2x upsample, align_corners=True =================
__global__ void upsample2_kernel(const float* x, float* y, int H, int W, int total)
{
    int i = blockIdx.x * blockDim.x + threadIdx.x;
    if (i >= total) return;
    int H2 = 2 * H, W2 = 2 * W;
    int bc  = i / (H2 * W2);
    int rem = i % (H2 * W2);
    int oy = rem / W2, ox = rem % W2;
    float fy = (float)oy * ((float)(H - 1) / (float)(H2 - 1));
    float fx = (float)ox * ((float)(W - 1) / (float)(W2 - 1));
    int y0 = (int)floorf(fy); int y1 = min(y0 + 1, H - 1); float ty = fy - (float)y0;
    int x0 = (int)floorf(fx); int x1 = min(x0 + 1, W - 1); float tx = fx - (float)x0;
    const float* p = x + (size_t)bc * H * W;
    float v00 = p[(size_t)y0 * W + x0], v01 = p[(size_t)y0 * W + x1];
    float v10 = p[(size_t)y1 * W + x0], v11 = p[(size_t)y1 * W + x1];
    float top = v00 * (1.f - tx) + v01 * tx;
    float bot = v10 * (1.f - tx) + v11 * tx;
    y[i] = top * (1.f - ty) + bot * ty;
}

// ================= area attention (aw = 4) =================
__global__ void attn_row_kernel(const float* q, const float* k, const float* v,
                                float* out, int C, int c8, int H, int W, int total)
{
    int i = blockIdx.x * blockDim.x + threadIdx.x;
    if (i >= total) return;
    const int nH = H / 4;
    int b   = i / (nH * W);
    int rem = i % (nH * W);
    int n   = rem / W;
    int x   = rem % W;
    int y0  = n * 4;
    float L[4][4];
#pragma unroll
    for (int a = 0; a < 4; ++a)
#pragma unroll
        for (int kk = 0; kk < 4; ++kk) L[a][kk] = 0.f;
    for (int c = 0; c < c8; ++c) {
        const float* qp = q + (((size_t)b * c8 + c) * H + y0) * W + x;
        const float* kp = k + (((size_t)b * c8 + c) * H + y0) * W + x;
        float qv[4], kv[4];
#pragma unroll
        for (int a = 0; a < 4; ++a) { qv[a] = qp[(size_t)a * W]; kv[a] = kp[(size_t)a * W]; }
#pragma unroll
        for (int a = 0; a < 4; ++a)
#pragma unroll
            for (int kk = 0; kk < 4; ++kk) L[a][kk] += qv[a] * kv[kk];
    }
#pragma unroll
    for (int a = 0; a < 4; ++a) {
        float mx = fmaxf(fmaxf(L[a][0], L[a][1]), fmaxf(L[a][2], L[a][3]));
        float s = 0.f;
#pragma unroll
        for (int kk = 0; kk < 4; ++kk) { L[a][kk] = expf(L[a][kk] - mx); s += L[a][kk]; }
        float inv = 1.f / s;
#pragma unroll
        for (int kk = 0; kk < 4; ++kk) L[a][kk] *= inv;
    }
    for (int cc = 0; cc < C; ++cc) {
        const float* vp = v + (((size_t)b * C + cc) * H + y0) * W + x;
        float* op = out + (((size_t)b * C + cc) * H + y0) * W + x;
        float vv[4];
#pragma unroll
        for (int kk = 0; kk < 4; ++kk) vv[kk] = vp[(size_t)kk * W];
#pragma unroll
        for (int a = 0; a < 4; ++a) {
            float o = 0.f;
#pragma unroll
            for (int kk = 0; kk < 4; ++kk) o += vv[kk] * L[a][kk];
            op[(size_t)a * W] = o;
        }
    }
}

__global__ void attn_col_kernel(const float* q, const float* k, const float* v,
                                float* out, int C, int c8, int H, int W, int total)
{
    int i = blockIdx.x * blockDim.x + threadIdx.x;
    if (i >= total) return;
    const int nW = W / 4;
    int b   = i / (H * nW);
    int rem = i % (H * nW);
    int h   = rem / nW;
    int n   = rem % nW;
    int x0  = n * 4;
    float L[4][4];
#pragma unroll
    for (int a = 0; a < 4; ++a)
#pragma unroll
        for (int kk = 0; kk < 4; ++kk) L[a][kk] = 0.f;
    for (int c = 0; c < c8; ++c) {
        const float* qp = q + (((size_t)b * c8 + c) * H + h) * W + x0;
        const float* kp = k + (((size_t)b * c8 + c) * H + h) * W + x0;
        float qv[4], kv[4];
#pragma unroll
        for (int a = 0; a < 4; ++a) { qv[a] = qp[a]; kv[a] = kp[a]; }
#pragma unroll
        for (int a = 0; a < 4; ++a)
#pragma unroll
            for (int kk = 0; kk < 4; ++kk) L[a][kk] += qv[a] * kv[kk];
    }
#pragma unroll
    for (int a = 0; a < 4; ++a) {
        float mx = fmaxf(fmaxf(L[a][0], L[a][1]), fmaxf(L[a][2], L[a][3]));
        float s = 0.f;
#pragma unroll
        for (int kk = 0; kk < 4; ++kk) { L[a][kk] = expf(L[a][kk] - mx); s += L[a][kk]; }
        float inv = 1.f / s;
#pragma unroll
        for (int kk = 0; kk < 4; ++kk) L[a][kk] *= inv;
    }
    for (int cc = 0; cc < C; ++cc) {
        const float* vp = v + (((size_t)b * C + cc) * H + h) * W + x0;
        float* op = out + (((size_t)b * C + cc) * H + h) * W + x0;
        float vv[4];
#pragma unroll
        for (int kk = 0; kk < 4; ++kk) vv[kk] = vp[kk];
#pragma unroll
        for (int a = 0; a < 4; ++a) {
            float o = 0.f;
#pragma unroll
            for (int kk = 0; kk < 4; ++kk) o += vv[kk] * L[a][kk];
            op[a] = o;
        }
    }
}

__global__ void attn_combine_kernel(const float* hA, const float* vA, const float* x,
                                    const float* gamma, float* out, int total)
{
    int i = blockIdx.x * blockDim.x + threadIdx.x;
    if (i >= total) return;
    out[i] = gamma[0] * (hA[i] + vA[i]) + x[i];
}

// =====================================================================
extern "C" void kernel_launch(void* const* d_in, const int* in_sizes, int n_in,
                              void* d_out, int out_size, void* d_ws, size_t ws_size,
                              hipStream_t stream)
{
    (void)in_sizes; (void)n_in; (void)out_size; (void)ws_size;
    const int B = 8;
    const int H0 = 320, W0 = 320; const int HW0 = H0 * W0;
    const int H1 = 160, W1 = 160; const int HW1 = H1 * W1;
    const int H2 = 80,  W2 = 80;  const int HW2 = H2 * W2;
    const int f0 = 24, f1 = 48, f2 = 96;

    auto P = [&](int i) { return (const float*)d_in[i]; };
    const float* X = P(0);

    float* ws    = (float*)d_ws;
    float* S1    = ws;
    float* S2    = ws + 1024;
    float* mBv   = ws + 2048;
    float* vBv   = ws + 2176;
    float* sca   = ws + 2304;
    _Float16* wpk = (_Float16*)(ws + 4096);                 // 32768 halves
    _Float16* apk = (_Float16*)(ws + 4096 + 16384);         // up to 78,643,200 halves
    float* arena = ws + 4096 + 16384 + 39321600;

    float* x0buf   = arena;                             // B*f1*HW0 (skip to dec2)
    float* x1buf   = x0buf + (size_t)B * f1 * HW0;      // B*f2*HW1 (skip to dec1)
    float* scratch = x1buf + (size_t)B * f2 * HW1;

    size_t cur = 0;
    auto A = [&](size_t n) { float* p = scratch + cur; cur += n; return p; };

    dim3 blk(256);
    auto EW = [&](int total) { return dim3((unsigned)((total + 255) / 256)); };

    auto packW = [&](const float* w, int Cout, int K, int Kpad, int Mpad) {
        int total = Mpad * Kpad;
        pack_w_kernel<<<EW(total), blk, 0, stream>>>(w, wpk, Cout, K, Kpad, total);
    };
    auto packA = [&](const float* i0, int C0, const float* i1, int C1,
                     const float* i2, int C2, int Kpad, int HW, int relu) {
        int K = C0 + C1 + C2;
        dim3 g((unsigned)(HW / 64), (unsigned)(Kpad / 32), (unsigned)B);
        pack_act_kernel<<<g, blk, 0, stream>>>(i0, C0, i1, C1, i2, C2, apk, K, Kpad, HW, relu);
    };
    auto gemmP = [&](const float* bias, const float* res, float* out,
                     int Cout, int HW, int Kpad, int rout) {
        int Mpad = ((Cout + 15) / 16) * 16;
        dim3 g((unsigned)(HW / 256), (unsigned)(Mpad / 16), (unsigned)B);
        gemm_packed_kernel<<<g, blk, 0, stream>>>(wpk, apk, bias, res, out, Cout, HW, Kpad, rout);
    };
    auto gemm = [&](const float* i0, int C0, const float* i1, int C1,
                    const float* i2, int C2, const float* w, const float* bias,
                    const float* res, float* out, int Cout, int HW,
                    int rin, int rout) {
        int K = C0 + C1 + C2;
        int Kpad = ((K + 31) / 32) * 32;
        int Mpad = ((Cout + 15) / 16) * 16;
        packW(w, Cout, K, Kpad, Mpad);
        packA(i0, C0, i1, C1, i2, C2, Kpad, HW, rin);
        gemmP(bias, res, out, Cout, HW, Kpad, rout);
    };
    auto stats = [&](const float* x, int C, int HW) {
        reduce_stats_kernel<<<dim3((unsigned)(B * C)), blk, 0, stream>>>(x, S1, S2, HW);
    };
    auto bstats = [&](int C, int HW) {
        batch_stats_kernel<<<dim3(1), dim3(128), 0, stream>>>(S1, S2, mBv, vBv, B, C, HW);
    };
    auto hnorm = [&](const float* x, float* y, const float* gw, const float* gb, int C, int HW) {
        stats(x, C, HW); bstats(C, HW);
        int total = B * C * HW;
        hnorm_apply_kernel<<<EW(total), blk, 0, stream>>>(x, y, S1, S2, mBv, vBv, gw, gb, C, HW, total);
    };
    auto ca = [&](float* x, const float* w1, const float* w2, int C, int Ch, int HW) {
        stats(x, C, HW);
        ca_compute_kernel<<<dim3((unsigned)B), dim3(128), 0, stream>>>(S1, w1, w2, sca, C, Ch, HW);
        int total = B * C * HW;
        ca_scale_kernel<<<EW(total), blk, 0, stream>>>(x, x, sca, HW, total);
    };
    auto axial_dw = [&](const float* x, const float* hw, const float* hb,
                        const float* vw, const float* vb, float* y, int C, int H, int W, int k) {
        int total = B * C * H * W;
        axial_dw_kernel<<<EW(total), blk, 0, stream>>>(x, hw, hb, vw, vb, y, C, H, W, k, total);
    };
    // area attention: pack x once, reuse packed operand for q/k/v projections
    auto area_attn = [&](const float* x, int C, int H, int W,
                         const float* qw, const float* qb, const float* kw, const float* kb,
                         const float* vw, const float* vb, const float* gamma,
                         float* qbuf, float* kbuf, float* vbuf, float* hA, float* vA, float* out) {
        int c8 = C / 8; int HW = H * W;
        int Kpad = ((C + 31) / 32) * 32;
        packA(x, C, nullptr, 0, nullptr, 0, Kpad, HW, 0);
        packW(qw, c8, C, Kpad, 16); gemmP(qb, nullptr, qbuf, c8, HW, Kpad, 0);
        packW(kw, c8, C, Kpad, 16); gemmP(kb, nullptr, kbuf, c8, HW, Kpad, 0);
        packW(vw, C,  C, Kpad, ((C + 15) / 16) * 16); gemmP(vb, nullptr, vbuf, C, HW, Kpad, 0);
        int tr = B * (H / 4) * W;
        attn_row_kernel<<<EW(tr), blk, 0, stream>>>(qbuf, kbuf, vbuf, hA, C, c8, H, W, tr);
        int tc = B * H * (W / 4);
        attn_col_kernel<<<EW(tc), blk, 0, stream>>>(qbuf, kbuf, vbuf, vA, C, c8, H, W, tc);
        int total = B * C * HW;
        attn_combine_kernel<<<EW(total), blk, 0, stream>>>(hA, vA, x, gamma, out, total);
    };

    // ---------------- Phase A: in-conv + enc1 @320 ----------------
    float* p1 = A((size_t)B * f1 * HW1);
    size_t mark1 = cur;
    float* t0  = A((size_t)B * f0 * HW0);
    float* axt = A((size_t)B * f0 * HW0);
    float* pw1 = A((size_t)B * f1 * HW0);
    {
        int total = B * f0 * HW0;
        conv3x3_kernel<<<EW(total), blk, 0, stream>>>(X, P(1), P(2), t0, 3, f0, H0, W0, total);
    }
    axial_dw(t0, P(5), P(6), P(7), P(8), axt, f0, H0, W0, 5);
    gemm(axt, f0, nullptr, 0, nullptr, 0, P(9), P(10), nullptr, pw1, f1, HW0, 0, 0);
    hnorm(pw1, x0buf, P(11), P(12), f1, HW0);
    ca(x0buf, P(13), P(14), f1, 3, HW0);
    {
        int total = B * f1 * HW1;
        maxpool2_kernel<<<EW(total), blk, 0, stream>>>(x0buf, p1, H0, W0, total);
    }
    cur = mark1;

    // ---------------- Phase B: enc2 @160 ----------------
    float* p2 = A((size_t)B * f2 * HW2);
    size_t mark2 = cur;
    float* axt2  = A((size_t)B * f1 * HW1);
    float* pw2   = A((size_t)B * f2 * HW1);
    float* vbufB = A((size_t)B * f2 * HW1);
    float* hAB   = A((size_t)B * f2 * HW1);
    float* vAB   = A((size_t)B * f2 * HW1);
    axial_dw(p1, P(15), P(16), P(17), P(18), axt2, f1, H1, W1, 5);
    gemm(axt2, f1, nullptr, 0, nullptr, 0, P(19), P(20), nullptr, pw2, f2, HW1, 0, 0);
    hnorm(pw2, pw2, P(21), P(22), f2, HW1);
    float* qB = axt2;                              // axt2 dead -> reuse for q,k
    float* kB = axt2 + (size_t)B * (f2 / 8) * HW1;
    area_attn(pw2, f2, H1, W1, P(23), P(24), P(25), P(26), P(27), P(28), P(29),
              qB, kB, vbufB, hAB, vAB, x1buf);
    {
        int total = B * f2 * HW2;
        maxpool2_kernel<<<EW(total), blk, 0, stream>>>(x1buf, p2, H1, W1, total);
    }
    cur = mark2;

    // ---------------- Phase C: bottleneck @80 ----------------
    float* bx = A((size_t)B * f2 * HW2);
    size_t mark3 = cur;
    float* dwt = A((size_t)B * f2 * HW2);
    float* y1  = A((size_t)B * f2 * HW2);
    float* y2  = A((size_t)B * f2 * HW2);
    float* y3  = A((size_t)B * f2 * HW2);
    float* fb  = A((size_t)B * f2 * HW2);
    axial_dw(p2, P(30), P(31), P(32), P(33), dwt, f2, H2, W2, 3);
    gemm(dwt, f2, nullptr, 0, nullptr, 0, P(34), P(35), nullptr, y1, f2, HW2, 0, 0);
    axial_dw(p2, P(36), P(37), P(38), P(39), dwt, f2, H2, W2, 5);
    gemm(dwt, f2, nullptr, 0, nullptr, 0, P(40), P(41), nullptr, y2, f2, HW2, 0, 0);
    axial_dw(p2, P(42), P(43), P(44), P(45), dwt, f2, H2, W2, 7);
    gemm(dwt, f2, nullptr, 0, nullptr, 0, P(46), P(47), nullptr, y3, f2, HW2, 0, 0);
    gemm(y1, f2, y2, f2, y3, f2, P(48), P(49), nullptr, fb, f2, HW2, 0, 0);  // fuse concat
    stats(fb, f2, HW2); bstats(f2, HW2);
    {
        int total = B * f2 * HW2;
        bn_apply_kernel<<<EW(total), blk, 0, stream>>>(fb, fb, mBv, vBv, P(50), P(51), f2, HW2, total, 1);
    }
    ca(fb, P(52), P(53), f2, 6, HW2);
    gemm(fb, f2, nullptr, 0, nullptr, 0, P(54), P(55), p2, bx, f2, HW2, 1, 0);  // +idn
    cur = mark3;

    // ---------------- Phase D: dec1 @160 ----------------
    float* c2buf = A((size_t)B * f1 * HW1);
    size_t mark4 = cur;
    float* up1 = A((size_t)B * f2 * HW1);
    float* c1  = A((size_t)B * f2 * HW1);
    float* hA2 = A((size_t)B * f1 * HW1);
    float* vA2 = A((size_t)B * f1 * HW1);
    {
        int total = B * f2 * HW1;
        upsample2_kernel<<<EW(total), blk, 0, stream>>>(bx, up1, H2, W2, total);
    }
    gemm(x1buf, f2, up1, f2, nullptr, 0, P(56), P(57), nullptr, c1, f2, HW1, 0, 0);
    float* dwt2 = up1;   // reuse (c1 gemm done with up1)
    axial_dw(c1, P(58), P(59), P(60), P(61), dwt2, f2, H1, W1, 5);
    float* pwd = c1;     // reuse (reads dwt2, writes pwd)
    gemm(dwt2, f2, nullptr, 0, nullptr, 0, P(62), P(63), nullptr, pwd, f1, HW1, 0, 0);
    float* q2 = up1;     // dwt2 dead -> q,k,v in its slot
    float* k2 = up1 + (size_t)B * (f1 / 8) * HW1;
    float* v2 = up1 + (size_t)2 * B * (f1 / 8) * HW1;
    area_attn(pwd, f1, H1, W1, P(64), P(65), P(66), P(67), P(68), P(69), P(70),
              q2, k2, v2, hA2, vA2, hA2);
    gemm(hA2, f1, nullptr, 0, nullptr, 0, P(71), P(72), nullptr, c2buf, f1, HW1, 1, 1);
    cur = mark4;

    // ---------------- Phase E: dec2 @320 + out ----------------
    float* up2 = A((size_t)B * f1 * HW0);
    float* c1e = A((size_t)B * f1 * HW0);
    {
        int total = B * f1 * HW0;
        upsample2_kernel<<<EW(total), blk, 0, stream>>>(c2buf, up2, H1, W1, total);
    }
    gemm(x0buf, f1, up2, f1, nullptr, 0, P(73), P(74), nullptr, c1e, f1, HW0, 0, 0);
    float* axt3 = up2;   // reuse
    axial_dw(c1e, P(75), P(76), P(77), P(78), axt3, f1, H0, W0, 5);
    float* pwe = c1e;    // reuse (reads axt3, writes pwe)
    gemm(axt3, f1, nullptr, 0, nullptr, 0, P(79), P(80), nullptr, pwe, f0, HW0, 0, 0);
    hnorm(pwe, pwe, P(81), P(82), f0, HW0);
    ca(pwe, P(83), P(84), f0, 1, HW0);
    float* c2e = up2;    // reuse (axt3 dead)
    gemm(pwe, f0, nullptr, 0, nullptr, 0, P(85), P(86), nullptr, c2e, f0, HW0, 1, 1);
    gemm(c2e, f0, nullptr, 0, nullptr, 0, P(3), P(4), nullptr, (float*)d_out, 3, HW0, 0, 0);
}
